// MyModel_31095563223116
// MI455X (gfx1250) — matrix operand order repaired and checked by reading, not compiled, not run on hardware
//
#include <hip/hip_runtime.h>
#include <stdint.h>

typedef __attribute__((ext_vector_type(16))) _Float16 v16h;
typedef __attribute__((ext_vector_type(8)))  _Float16 v8h;
typedef __attribute__((ext_vector_type(4)))  _Float16 v4h;
typedef __attribute__((ext_vector_type(2)))  _Float16 v2h;
typedef __attribute__((ext_vector_type(8)))  float    v8f;

#define HID 128
#define OUTD 64

// Packed f16 atomic-add path (GLOBAL_ATOMIC_PK_ADD_F16) when the builtin is
// available for gfx1250; otherwise fall back to f32 atomics.
#if defined(__has_builtin)
#if __has_builtin(__builtin_amdgcn_global_atomic_fadd_v2f16)
#define AGG_F16 1
#endif
#endif
#ifndef AGG_F16
#define AGG_F16 0
#endif

#if AGG_F16
typedef _Float16 AggT;
typedef __attribute__((address_space(1))) v2h* gptr_v2h;
#else
typedef float AggT;
#endif

// ---------------------------------------------------------------- utilities
__global__ void k_zero(float* __restrict__ p, long long n) {
  long long i = (long long)blockIdx.x * blockDim.x + threadIdx.x;
  long long stride = (long long)gridDim.x * blockDim.x;
  for (; i < n; i += stride) p[i] = 0.f;
}

__global__ void k_deg_acc(const int* __restrict__ dst, float* __restrict__ deg, int E) {
  int e = blockIdx.x * blockDim.x + threadIdx.x;
  if (e < E) atomicAdd(&deg[dst[e]], 1.0f);
}

__global__ void k_deg_inv(float* __restrict__ deg, int N) {
  int i = blockIdx.x * blockDim.x + threadIdx.x;
  if (i < N) {
    float v = deg[i];
    deg[i] = 1.0f / (v > 1.0f ? v : 1.0f);
  }
}

// ------------------------------------------------------------- aggregation
// layer-1 aggregation of raw x features (IN = 8), f32 atomics
__global__ void k_agg_x(const float* __restrict__ x, const int* __restrict__ src,
                        const int* __restrict__ dst, float* __restrict__ agg, int E) {
  int e = blockIdx.x * blockDim.x + threadIdx.x;
  if (e >= E) return;
  int s = src[e], d = dst[e];
#pragma unroll
  for (int k = 0; k < 8; k++) atomicAdd(&agg[(size_t)d * 8 + k], x[(size_t)s * 8 + k]);
}

// mid-layer aggregation: gather f16 rows (L2-resident); scatter via packed
// f16 atomics when available (halves atomic count + bytes), else f32 atomics.
// one wave handles one edge: lane c covers features 4c..4c+3
__global__ void k_agg_h(const _Float16* __restrict__ h, const int* __restrict__ src,
                        const int* __restrict__ dst, AggT* __restrict__ agg, int E) {
  long long t = (long long)blockIdx.x * blockDim.x + threadIdx.x;
  int e = (int)(t >> 5);
  int c = ((int)t & 31) * 4;
  if (e >= E) return;
  int s = src[e], d = dst[e];
  v4h hv = *(const v4h*)(h + (size_t)s * HID + c);
  AggT* ap = agg + (size_t)d * HID + c;
#if AGG_F16
  v2h p0; p0[0] = hv[0]; p0[1] = hv[1];
  v2h p1; p1[0] = hv[2]; p1[1] = hv[3];
  __builtin_amdgcn_global_atomic_fadd_v2f16((gptr_v2h)(uintptr_t)(ap), p0);
  __builtin_amdgcn_global_atomic_fadd_v2f16((gptr_v2h)(uintptr_t)(ap + 2), p1);
#else
#pragma unroll
  for (int q = 0; q < 4; q++) atomicAdd(ap + q, (float)hv[q]);
#endif
}

// ----------------------------------------------------- layer 1 (K=8, scalar)
__global__ __launch_bounds__(256) void k_sage_first(
    const float* __restrict__ x, const float* __restrict__ aggx,
    const float* __restrict__ deginv, const float* __restrict__ w1s,
    const float* __restrict__ w1n, const float* __restrict__ b1,
    _Float16* __restrict__ h_out, int N) {
  __shared__ float Ws[8 * HID];
  __shared__ float Wn[8 * HID];
  __shared__ float Bb[HID];
  const int tid = threadIdx.x;
  for (int idx = tid; idx < 8 * HID; idx += 256) { Ws[idx] = w1s[idx]; Wn[idx] = w1n[idx]; }
  if (tid < HID) Bb[tid] = b1[tid];
  __syncthreads();
  const int gi = blockIdx.x * 256 + tid;
  const int i = gi >> 7, j = gi & 127;
  if (i >= N) return;
  const float dv = deginv[i];
  float acc = Bb[j];
#pragma unroll
  for (int k = 0; k < 8; k++)
    acc += x[(size_t)i * 8 + k] * Ws[k * HID + j] + aggx[(size_t)i * 8 + k] * dv * Wn[k * HID + j];
  h_out[(size_t)i * HID + j] = (_Float16)(acc > 0.f ? acc : 0.f);
}

// ------------------------------------------- fused SAGE GEMM (WMMA, K = 256)
// h_out[N,NCOL] = relu([h_in | agg*deginv] @ [Ws;Wn] + b), f16 in / f32 acc
template <int NCOL>
__global__ __launch_bounds__(256) void k_sage_wmma(
    const _Float16* __restrict__ h_in, const AggT* __restrict__ agg,
    const float* __restrict__ deginv, const float* __restrict__ ws,
    const float* __restrict__ wn, const float* __restrict__ bias,
    _Float16* __restrict__ h_out, int N) {
  __shared__ _Float16 Bs[256 * NCOL];  // [Ws;Wn] as f16
  const int tid = threadIdx.x;
  for (int idx = tid; idx < 256 * NCOL; idx += 256) {
    int kk = idx / NCOL, n = idx % NCOL;
    float v = (kk < HID) ? ws[kk * NCOL + n] : wn[(kk - HID) * NCOL + n];
    Bs[idx] = (_Float16)v;
  }
  __syncthreads();

  const int wave = tid >> 5, lane = tid & 31;
  const int hi = lane >> 4, lm = lane & 15;
  const int m_base = blockIdx.x * 128 + wave * 16;
  int mA = m_base + lm;
  if (mA >= N) mA = N - 1;  // clamp loads; stores are guarded (EXEC stays full for WMMA)
  const _Float16* hrow = h_in + (size_t)mA * HID;
  const AggT* arow = agg + (size_t)mA * HID;
  const float dv = deginv[mA];

  constexpr int NT = NCOL / 16;
  v8f acc[NT];
#pragma unroll
  for (int t = 0; t < NT; t++) acc[t] = {};

#pragma unroll
  for (int ks = 0; ks < 8; ks++) {
    const int k0 = ks * 32;
    v16h a;
    if (ks < 4) {  // self part: direct f16 rows, two contiguous 8-half runs per lane
      v8h r0 = *(const v8h*)(hrow + k0 + hi * 8);
      v8h r1 = *(const v8h*)(hrow + k0 + 16 + hi * 8);
#pragma unroll
      for (int j = 0; j < 8; j++) { a[j] = r0[j]; a[8 + j] = r1[j]; }
    } else {  // neighbor part: agg * deg_inv -> f16
      const int q0 = k0 - HID;
#if AGG_F16
      v8h r0 = *(const v8h*)(arow + q0 + hi * 8);
      v8h r1 = *(const v8h*)(arow + q0 + 16 + hi * 8);
#pragma unroll
      for (int j = 0; j < 8; j++) {
        a[j]     = (_Float16)((float)r0[j] * dv);
        a[8 + j] = (_Float16)((float)r1[j] * dv);
      }
#else
#pragma unroll
      for (int j = 0; j < 8; j++) {
        a[j]     = (_Float16)(arow[q0 + hi * 8 + j] * dv);
        a[8 + j] = (_Float16)(arow[q0 + 16 + hi * 8 + j] * dv);
      }
#endif
    }
    const _Float16* brow = Bs + k0 * NCOL + lm;  // B-frag: lane=n, k in table order
#pragma unroll
    for (int t = 0; t < NT; t++) {
      v16h b;
#pragma unroll
      for (int j = 0; j < 8; j++) {
        b[j]     = brow[(hi * 8 + j) * NCOL + t * 16];
        b[8 + j] = brow[(16 + hi * 8 + j) * NCOL + t * 16];
      }
      acc[t] = __builtin_amdgcn_wmma_f32_16x16x32_f16(false, a, false, b, (short)0,
                                                      acc[t], false, false);
    }
  }

#pragma unroll
  for (int t = 0; t < NT; t++) {
    const int n = t * 16 + lm;
    const float bv = bias[n];
#pragma unroll
    for (int r = 0; r < 8; r++) {
      const int m = m_base + r + 8 * hi;  // C layout: VGPR r -> M=r (+8 for hi lanes)
      if (m < N) {
        float v = acc[t][r] + bv;
        h_out[(size_t)m * NCOL + n] = (_Float16)(v > 0.f ? v : 0.f);
      }
    }
  }
}

// ------------------------------------------------- edge MLP (WMMA + VALU)
// per wave: 16 edges. layer0 (128->20) via WMMA; 3x(20->20)+20->8 via VALU.
__global__ __launch_bounds__(256) void k_edge_mlp(
    const _Float16* __restrict__ h8, const int* __restrict__ src,
    const int* __restrict__ dst, const float* __restrict__ mw0,
    const float* __restrict__ mb0, const float* __restrict__ mwh,
    const float* __restrict__ mbh, const float* __restrict__ mwf,
    const float* __restrict__ mbf, float* __restrict__ out, int E) {
  __shared__ _Float16 W0s[128 * 32];      // mw0 padded 20->32 cols
  __shared__ float Whs[3 * 20 * 20];
  __shared__ float Wfs[20 * 8];
  __shared__ float B0s[32];
  __shared__ float Bhs[3 * 20];
  __shared__ float Bfs[8];
  __shared__ float Es[8][2][16][20];      // per-wave ping-pong activations

  const int tid = threadIdx.x;
  for (int idx = tid; idx < 128 * 32; idx += 256) {
    int k = idx >> 5, n = idx & 31;
    W0s[idx] = (_Float16)((n < 20) ? mw0[k * 20 + n] : 0.f);
  }
  for (int idx = tid; idx < 1200; idx += 256) Whs[idx] = mwh[idx];
  if (tid < 160) Wfs[tid] = mwf[tid];
  if (tid < 32) B0s[tid] = (tid < 20) ? mb0[tid] : 0.f;
  if (tid < 60) Bhs[tid] = mbh[tid];
  if (tid < 8) Bfs[tid] = mbf[tid];
  __syncthreads();

  const int wave = tid >> 5, lane = tid & 31;
  const int hi = lane >> 4, lm = lane & 15;
  int e = blockIdx.x * 128 + wave * 16 + lm;
  if (e >= E) e = E - 1;  // clamp (grid divides E exactly here anyway)
  const int s = src[e], d = dst[e];
  const _Float16* hs = h8 + (size_t)s * OUTD;
  const _Float16* hd = h8 + (size_t)d * OUTD;

  v8f acc0 = {}, acc1 = {};
#pragma unroll
  for (int ks = 0; ks < 4; ks++) {
    const int k0 = ks * 32;
    const int ka = k0 + hi * 8;
    const int kb = k0 + 16 + hi * 8;
    const _Float16* pa = (ka < OUTD) ? (hs + ka) : (hd + ka - OUTD);
    const _Float16* pb = (kb < OUTD) ? (hs + kb) : (hd + kb - OUTD);
    v8h r0 = *(const v8h*)pa;
    v8h r1 = *(const v8h*)pb;
    v16h a;
#pragma unroll
    for (int j = 0; j < 8; j++) { a[j] = r0[j]; a[8 + j] = r1[j]; }
    const _Float16* brow = W0s + k0 * 32 + lm;
    v16h b0, b1;
#pragma unroll
    for (int j = 0; j < 8; j++) {
      b0[j] = brow[(hi * 8 + j) * 32]; b0[8 + j] = brow[(16 + hi * 8 + j) * 32];
      b1[j] = brow[(hi * 8 + j) * 32 + 16]; b1[8 + j] = brow[(16 + hi * 8 + j) * 32 + 16];
    }
    acc0 = __builtin_amdgcn_wmma_f32_16x16x32_f16(false, a, false, b0, (short)0, acc0, false, false);
    acc1 = __builtin_amdgcn_wmma_f32_16x16x32_f16(false, a, false, b1, (short)0, acc1, false, false);
  }
  // scatter e1 = relu(acc + b) into per-wave LDS (cols 0..19 valid)
#pragma unroll
  for (int r = 0; r < 8; r++) {
    const int m = r + 8 * hi;
    float v0 = acc0[r] + B0s[lm];
    Es[wave][0][m][lm] = v0 > 0.f ? v0 : 0.f;
    const int n1 = 16 + lm;
    if (n1 < 20) {
      float v1 = acc1[r] + B0s[n1];
      Es[wave][0][m][n1] = v1 > 0.f ? v1 : 0.f;
    }
  }
  // hidden 20x20 layers: 2 lanes per edge (same-wave LDS is in-order)
  const int em = lane >> 1, half = lane & 1;
  int cur = 0;
#pragma unroll
  for (int t = 0; t < 3; t++) {
    float o[10];
#pragma unroll
    for (int j = 0; j < 10; j++) {
      const int jj = half * 10 + j;
      float sa = Bhs[t * 20 + jj];
#pragma unroll
      for (int k = 0; k < 20; k++) sa += Es[wave][cur][em][k] * Whs[t * 400 + k * 20 + jj];
      o[j] = sa > 0.f ? sa : 0.f;
    }
#pragma unroll
    for (int j = 0; j < 10; j++) Es[wave][cur ^ 1][em][half * 10 + j] = o[j];
    cur ^= 1;
  }
  // final 20->8
  const int eo = blockIdx.x * 128 + wave * 16 + em;
  if (eo < E) {
#pragma unroll
    for (int j = 0; j < 4; j++) {
      const int jj = half * 4 + j;
      float sa = Bfs[jj];
#pragma unroll
      for (int k = 0; k < 20; k++) sa += Es[wave][cur][em][k] * Wfs[k * 8 + jj];
      out[(size_t)eo * 8 + jj] = sa;
    }
  }
}

// ------------------------------------------------------------------ launch
extern "C" void kernel_launch(void* const* d_in, const int* in_sizes, int n_in,
                              void* d_out, int out_size, void* d_ws, size_t ws_size,
                              hipStream_t stream) {
  const float* x   = (const float*)d_in[0];
  const int* src   = (const int*)d_in[1];
  const int* dst   = (const int*)d_in[2];
  const float* w1s = (const float*)d_in[3];
  const float* w1n = (const float*)d_in[4];
  const float* b1  = (const float*)d_in[5];
  const float* wms = (const float*)d_in[6];
  const float* wmn = (const float*)d_in[7];
  const float* bm  = (const float*)d_in[8];
  const float* wos = (const float*)d_in[9];
  const float* won = (const float*)d_in[10];
  const float* bo  = (const float*)d_in[11];
  const float* mw0 = (const float*)d_in[12];
  const float* mb0 = (const float*)d_in[13];
  const float* mwh = (const float*)d_in[14];
  const float* mbh = (const float*)d_in[15];
  const float* mwf = (const float*)d_in[16];
  const float* mbf = (const float*)d_in[17];

  const int N = in_sizes[0] / 8;
  const int E = in_sizes[1];

  size_t off = 0;
  auto alloc = [&](size_t bytes) -> void* {
    off = (off + 255) & ~(size_t)255;
    void* q = (char*)d_ws + off;
    off += bytes;
    return q;
  };
  float* deginv  = (float*)alloc((size_t)N * sizeof(float));
  // agg buffer used as f32 (layer 1) and AggT (mid/out layers)
  float* aggf    = (float*)alloc((size_t)N * HID * sizeof(float));
  AggT* agg      = (AggT*)aggf;
  _Float16* hA   = (_Float16*)alloc((size_t)N * HID * sizeof(_Float16));
  _Float16* hB   = (_Float16*)alloc((size_t)N * HID * sizeof(_Float16));
  (void)ws_size;

  const int TB = 256;
  const int gN   = (N + TB - 1) / TB;
  const int gE   = (E + TB - 1) / TB;
  const int gEW  = (int)(((long long)E * 32 + TB - 1) / TB);
  const int gRow = (N + 127) / 128;
  const int gNJ  = (int)(((long long)N * HID + TB - 1) / TB);
  const int gEd  = (E + 127) / 128;
  const long long aggWords = (long long)N * HID * (long long)sizeof(AggT) / 4;

  // degree -> 1/max(deg,1)
  k_zero<<<1024, TB, 0, stream>>>(deginv, (long long)N);
  k_deg_acc<<<gE, TB, 0, stream>>>(dst, deginv, E);
  k_deg_inv<<<gN, TB, 0, stream>>>(deginv, N);

  // layer 1 (IN=8), f32 aggregation
  k_zero<<<1024, TB, 0, stream>>>(aggf, (long long)N * 8);
  k_agg_x<<<gE, TB, 0, stream>>>(x, src, dst, aggf, E);
  k_sage_first<<<gNJ, TB, 0, stream>>>(x, aggf, deginv, w1s, w1n, b1, hA, N);

  // 6 mid layers (128 -> 128), WMMA
  _Float16* hin = hA;
  _Float16* hout = hB;
  for (int i = 0; i < 6; i++) {
    k_zero<<<2048, TB, 0, stream>>>(aggf, aggWords);
    k_agg_h<<<gEW, TB, 0, stream>>>(hin, src, dst, agg, E);
    k_sage_wmma<HID><<<gRow, TB, 0, stream>>>(hin, agg, deginv,
                                              wms + (size_t)i * HID * HID,
                                              wmn + (size_t)i * HID * HID,
                                              bm + (size_t)i * HID, hout, N);
    _Float16* t = hin; hin = hout; hout = t;
  }

  // output layer (128 -> 64), WMMA
  k_zero<<<2048, TB, 0, stream>>>(aggf, aggWords);
  k_agg_h<<<gEW, TB, 0, stream>>>(hin, src, dst, agg, E);
  k_sage_wmma<OUTD><<<gRow, TB, 0, stream>>>(hin, agg, deginv, wos, won, bo, hout, N);

  // per-edge MLP (WMMA first layer + VALU tail)
  k_edge_mlp<<<gEd, TB, 0, stream>>>(hout, src, dst, mw0, mb0, mwh, mbh, mwf, mbf,
                                     (float*)d_out, E);
}
